// WeightedMaxAggregator_25967372271703
// MI455X (gfx1250) — matrix operand order repaired and checked by reading, not compiled, not run on hardware
//
#include <hip/hip_runtime.h>
#include <stdint.h>

// ---- CDNA5 (gfx1250) wave32 WMMA types --------------------------------------
typedef __bf16 bf16_t;
typedef __attribute__((ext_vector_type(16))) __bf16 v16bf;
typedef __attribute__((ext_vector_type(8)))  __bf16 v8bf;
typedef __attribute__((ext_vector_type(2)))  __bf16 v2bf;
typedef __attribute__((ext_vector_type(8)))  float  v8f;

#define S_LEN 512               // seq_len == n_items
#define E_DIM 64                // embedding size
#define NT    (S_LEN / 16)      // 32 n-tiles per batch

// Low 32 bits of a flat shared-aperture pointer are the LDS byte address
// (see ISA 10.2).
__device__ __forceinline__ uint32_t lds_addr32(const void* p) {
  return (uint32_t)(uintptr_t)p;
}

// Async global->LDS copy, tracked with ASYNCcnt (ISA 15.18.3 op 97).
__device__ __forceinline__ void async_ld_b64(uint32_t lds_off, const void* gaddr) {
  asm volatile("global_load_async_to_lds_b64 %0, %1, off"
               :: "v"(lds_off), "v"(gaddr) : "memory");
}
__device__ __forceinline__ void wait_async0() {
  asm volatile("s_wait_asynccnt 0x0" ::: "memory");
}

// -----------------------------------------------------------------------------
// Kernel 1: L2-normalize each 64-float row of user_emb / item_emb, emit bf16.
// One wave per row: 32 lanes x float2 = 64 elements; butterfly reduce in-wave.
// -----------------------------------------------------------------------------
__global__ void __launch_bounds__(256)
wm_normalize(const float* __restrict__ u, const float* __restrict__ v,
             bf16_t* __restrict__ un, bf16_t* __restrict__ vn, int nrows) {
  const int lane = threadIdx.x & 31;
  const int wid  = blockIdx.x * (blockDim.x >> 5) + (threadIdx.x >> 5);
  if (wid >= nrows * 2) return;
  const bool isU = (wid < nrows);
  const int row  = isU ? wid : wid - nrows;

  const float2* rp = (const float2*)((isU ? u : v) + (size_t)row * E_DIM);
  float2 a = rp[lane];
  float ss = a.x * a.x + a.y * a.y;
#pragma unroll
  for (int off = 16; off > 0; off >>= 1) ss += __shfl_xor(ss, off, 32);
  const float inv = 1.0f / fmaxf(sqrtf(ss), 1e-7f);   // clamp(norm, 1e-7)

  v2bf o;
  o[0] = (bf16_t)(a.x * inv);
  o[1] = (bf16_t)(a.y * inv);
  *(v2bf*)((isU ? un : vn) + (size_t)row * E_DIM + lane * 2) = o;
}

// -----------------------------------------------------------------------------
// Kernel 2: fold cluster_weights * proj_w, proj_b and the binary mask into a
// per-(b,n) pair {scale, add}. Masked-out n -> {0, -inf}: fma(cos,0,-inf)=-inf,
// and sigmoid(-inf)=0 reproduces the reference's post-sigmoid mask exactly
// (including the all-masked row -> 0 case).
// -----------------------------------------------------------------------------
__global__ void wm_scale(const float* __restrict__ cw, const float* __restrict__ mask,
                         const float* __restrict__ pw, const float* __restrict__ pb,
                         float2* __restrict__ sa, int n) {
  const int i = blockIdx.x * blockDim.x + threadIdx.x;
  if (i >= n) return;
  const float w = pw[0], b = pb[0];
  const bool keep = (mask[i] != 0.0f);
  float2 o;
  o.x = keep ? cw[i] * w : 0.0f;
  o.y = keep ? b : -__builtin_inff();
  sa[i] = o;
}

// -----------------------------------------------------------------------------
// Kernel 3: block = 8 waves on ONE batch; each wave owns two 16-wide s-tiles
// (B fragments loop-invariant in 32 VGPRs). Per n-tile, the 2 KB v-hat tile is
// staged into LDS once per block via global_load_async_to_lds_b64 with double
// buffering (s_wait_asynccnt 0 -> barrier -> issue next -> compute current),
// then each wave builds A fragments from LDS (ds_load_b128) and runs
// 4x v_wmma_f32_16x16x32_bf16. Epilogue: x = fma(cos, scale[n], add[n]),
// running max; cross-half shuffle; one sigmoid per output.
//
// Fragment layouts used here (wave32):
//   A 16x32 bf16 : lane half h, row M=lane%16; elem j -> K = h*16 + j
//                  (16 contiguous, loaded as two 8-elem chunks)
//   B 32x16 bf16 : col N=lane%16; elem j -> K = h*16 + j (16 contiguous)
//   C/D 16x16 f32: reg r -> M = h*8 + r, N = lane%16
// -----------------------------------------------------------------------------
__global__ void __launch_bounds__(256)
wm_gemm_max(const bf16_t* __restrict__ un, const bf16_t* __restrict__ vn,
            const float2* __restrict__ sa, float* __restrict__ out) {
  __shared__ bf16_t atile[2][16 * E_DIM];           // 2 x 2 KB double buffer

  const int tid   = threadIdx.x;
  const int lane  = tid & 31;
  const int wave  = tid >> 5;
  const int b     = blockIdx.x >> 1;                // 2 blocks per batch
  const int s0    = (blockIdx.x & 1) * 256 + wave * 32;
  const int half  = lane >> 4;
  const int m     = lane & 15;

  // Loop-invariant B fragments for two s-tiles (u-hat columns, K split by half).
  const bf16_t* ub0 = un + (size_t)(b * S_LEN + s0 + m) * E_DIM;
  const bf16_t* ub1 = ub0 + 16 * E_DIM;
  const v16bf bf00 = *(const v16bf*)(ub0 + 0  + half * 16);   // tile0, K 0..31
  const v16bf bf01 = *(const v16bf*)(ub0 + 32 + half * 16);   // tile0, K 32..63
  const v16bf bf10 = *(const v16bf*)(ub1 + 0  + half * 16);   // tile1, K 0..31
  const v16bf bf11 = *(const v16bf*)(ub1 + 32 + half * 16);   // tile1, K 32..63

  const bf16_t* vb  = vn + (size_t)b * S_LEN * E_DIM;
  const float2* sab = sa + (size_t)b * S_LEN;

  // Prologue: stage n-tile 0 (256 threads x b64 = 2 KB).
  async_ld_b64(lds_addr32(&atile[0][tid * 4]), vb + tid * 4);

  float rm0 = -__builtin_inff();
  float rm1 = -__builtin_inff();

  for (int nt = 0; nt < NT; ++nt) {
    wait_async0();        // own async issues for tile nt complete
    __syncthreads();      // all waves' writes for tile nt visible in LDS

    if (nt + 1 < NT)      // overlap: load tile nt+1 into the other buffer
      async_ld_b64(lds_addr32(&atile[(nt + 1) & 1][tid * 4]),
                   vb + (size_t)(nt + 1) * (16 * E_DIM) + tid * 4);

    // A fragments from LDS (broadcast across the 8 waves).
    const bf16_t* arow = &atile[nt & 1][m * E_DIM];
    const v8bf a00 = *(const v8bf*)(arow +  0 + half * 16);
    const v8bf a01 = *(const v8bf*)(arow +  8 + half * 16);
    const v8bf a10 = *(const v8bf*)(arow + 32 + half * 16);
    const v8bf a11 = *(const v8bf*)(arow + 40 + half * 16);
    const v16bf af0 = __builtin_shufflevector(a00, a01, 0,1,2,3,4,5,6,7,8,9,10,11,12,13,14,15);
    const v16bf af1 = __builtin_shufflevector(a10, a11, 0,1,2,3,4,5,6,7,8,9,10,11,12,13,14,15);

    v8f acc0 = {};
    v8f acc1 = {};
    acc0 = __builtin_amdgcn_wmma_f32_16x16x32_bf16(false, af0, false, bf00,
                                                   (short)0, acc0, false, false);
    acc0 = __builtin_amdgcn_wmma_f32_16x16x32_bf16(false, af1, false, bf01,
                                                   (short)0, acc0, false, false);
    acc1 = __builtin_amdgcn_wmma_f32_16x16x32_bf16(false, af0, false, bf10,
                                                   (short)0, acc1, false, false);
    acc1 = __builtin_amdgcn_wmma_f32_16x16x32_bf16(false, af1, false, bf11,
                                                   (short)0, acc1, false, false);

    // Per-accumulator n index: n = nt*16 + half*8 + r (shared by both acc sets).
#pragma unroll
    for (int r = 0; r < 8; ++r) {
      const float2 c = sab[nt * 16 + half * 8 + r];
      rm0 = fmaxf(rm0, fmaf(acc0[r], c.x, c.y));
      rm1 = fmaxf(rm1, fmaf(acc1[r], c.x, c.y));
    }
  }

  // Lanes L and L+16 hold the same s column with disjoint n halves.
  rm0 = fmaxf(rm0, __shfl_xor(rm0, 16, 32));
  rm1 = fmaxf(rm1, __shfl_xor(rm1, 16, 32));
  if (half == 0) {
    float* o = out + (size_t)b * S_LEN + s0 + m;
    o[0]  = 1.0f / (1.0f + expf(-rm0));
    o[16] = 1.0f / (1.0f + expf(-rm1));
  }
}

// -----------------------------------------------------------------------------
extern "C" void kernel_launch(void* const* d_in, const int* in_sizes, int n_in,
                              void* d_out, int out_size, void* d_ws, size_t ws_size,
                              hipStream_t stream) {
  const float* user = (const float*)d_in[0];   // [B,S,E]
  const float* item = (const float*)d_in[1];   // [B,S,E]
  const float* cw   = (const float*)d_in[2];   // [B,S,1]
  const float* mask = (const float*)d_in[3];   // [B,1,S]
  const float* pw   = (const float*)d_in[4];   // [1,1]
  const float* pb   = (const float*)d_in[5];   // [1]
  float* out = (float*)d_out;                  // [B,S]

  const int BS = in_sizes[2];                  // B*S
  const int Bn = BS / S_LEN;

  // Workspace layout: [un bf16 | vn bf16 | sa float2]  (~17 MB total)
  const size_t NB = (size_t)BS * E_DIM * sizeof(bf16_t);
  bf16_t* un = (bf16_t*)d_ws;
  bf16_t* vn = (bf16_t*)((char*)d_ws + NB);
  float2* sa = (float2*)((char*)d_ws + 2 * NB);

  const int rows2   = BS * 2;                      // u rows + v rows
  const int blocks1 = (rows2 + 7) / 8;             // 8 waves / 256-thread block
  wm_normalize<<<blocks1, 256, 0, stream>>>(user, item, un, vn, BS);

  wm_scale<<<(BS + 255) / 256, 256, 0, stream>>>(cw, mask, pw, pb, sa, BS);

  wm_gemm_max<<<Bn * 2, 256, 0, stream>>>(un, vn, sa, out);
}
